// L2MultiheadAttention_3350074491384
// MI455X (gfx1250) — matrix operand order repaired and checked by reading, not compiled, not run on hardware
//
#include <hip/hip_runtime.h>

// ---------------------------------------------------------------------------
// CDNA5 (gfx1250) WMMA types and helpers
// ---------------------------------------------------------------------------
typedef __bf16 v16bf __attribute__((ext_vector_type(16)));
typedef float  v8f   __attribute__((ext_vector_type(8)));
typedef unsigned int u32x4 __attribute__((ext_vector_type(4)));
typedef int          i32x4 __attribute__((ext_vector_type(4)));
typedef int          i32x8 __attribute__((ext_vector_type(8)));

#if defined(__has_builtin)
#  if __has_builtin(__builtin_amdgcn_tensor_load_to_lds)
#    define HAVE_TDM 1
#  endif
#endif
#ifndef HAVE_TDM
#  define HAVE_TDM 0
#endif

union FragU {
    v16bf v;
    uint4 q[2];
};

__device__ __forceinline__ v8f wmma_bf16(v16bf a, v16bf b, v8f c) {
    // (neg_a, A, neg_b, B, c_mod, C, reuse_a, reuse_b)
    return __builtin_amdgcn_wmma_f32_16x16x32_bf16(false, a, false, b, (short)0, c,
                                                   false, false);
}

__device__ __forceinline__ unsigned short f2bf(float f) {
    unsigned u = __float_as_uint(f);
    unsigned r = u + 0x7FFFu + ((u >> 16) & 1u);  // round-to-nearest-even
    return (unsigned short)(r >> 16);
}

__device__ __forceinline__ float bf2f(unsigned short h) {
    return __uint_as_float(((unsigned)h) << 16);
}

// Fragment loads from LDS rows stored [row][k] (16B-aligned rows).
// A layout (16x32, 16-bit): m = lane&15; elems 0..7 -> k=koff..koff+7,
// elems 8..15 -> k=koff+16..koff+23, koff = (lane>=16)?8:0 (+k-chunk).
__device__ __forceinline__ v16bf load_frag_a(const unsigned short* row, int koff) {
    FragU u;
    u.q[0] = *(const uint4*)(row + koff);
    u.q[1] = *(const uint4*)(row + koff + 16);
    return u.v;
}
// B layout (32x16, 16-bit): n = lane&15 (selects LDS row of B^T tile);
// elems 0..7 -> k=koff2/2..+7, elems 8..15 -> k=koff2/2+16..+23, koff2 = (lane>=16)?16:0.
__device__ __forceinline__ v16bf load_frag_b(const unsigned short* row, int koff2) {
    FragU u;
    u.q[0] = *(const uint4*)(row + koff2 / 2);
    u.q[1] = *(const uint4*)(row + koff2 / 2 + 16);
    return u.v;
}

__device__ __forceinline__ void tdm_wait0() {
#if defined(__has_builtin) && __has_builtin(__builtin_amdgcn_s_wait_tensorcnt)
    __builtin_amdgcn_s_wait_tensorcnt(0);
#else
    asm volatile("s_wait_tensorcnt 0x0" ::: "memory");
#endif
}

#if HAVE_TDM
// TDM: load a 2D tile of bf16 (tile0 elems per row x tile1 rows) from global
// (row stride stride0 elements) into LDS, padding each row from 32 dwords to
// 36 dwords (pad_interval code 4 = 32 dwords, pad_amount code 3 = 4 dwords)
// -> LDS row pitch 72 ushorts, matching the padded LDS tiles below.
__device__ __forceinline__ void tdm_load_2d_bf16(unsigned lds_off,
                                                 unsigned long long gaddr,
                                                 unsigned tile0, unsigned tile1,
                                                 unsigned long long stride0) {
    u32x4 g0 = { 1u,  // count=1, user flags 0, no gather
                 lds_off,
                 (unsigned)(gaddr & 0xFFFFFFFFu),
                 (unsigned)((gaddr >> 32) & 0x01FFFFFFu) | (2u << 30) };  // type=2
    const unsigned d0 = (1u << 16)      // data_size = 1 (2 bytes)
                      | (1u << 20)      // pad_enable
                      | (4u << 22)      // pad_interval: 32 dwords
                      | (3u << 25);     // pad_amount: 4 dwords
    i32x8 g1 = { (int)d0,
                 (int)((tile0 & 0xFFFFu) << 16),                 // tensor_dim0 lo
                 (int)(((tile1 & 0xFFFFu) << 16)),               // dim0 hi=0 | tensor_dim1 lo
                 (int)((tile0 & 0xFFFFu) << 16),                 // dim1 hi=0 | tile_dim0
                 (int)(tile1 & 0xFFFFu),                         // tile_dim1 | tile_dim2=0
                 (int)(unsigned)(stride0 & 0xFFFFFFFFu),         // tensor_dim0_stride lo
                 (int)(unsigned)((stride0 >> 32) & 0xFFFFu),     // stride0 hi | stride1 lo=0
                 0 };
    i32x4 z4 = { 0, 0, 0, 0 };
#if __clang_major__ >= 23
    i32x8 z8 = { 0, 0, 0, 0, 0, 0, 0, 0 };
    __builtin_amdgcn_tensor_load_to_lds(g0, g1, z4, z4, z8, 0);
#else
    __builtin_amdgcn_tensor_load_to_lds(g0, g1, z4, z4, 0);
#endif
}
#endif  // HAVE_TDM

// ---------------------------------------------------------------------------
// fp32 -> bf16 conversion (vectorized x4)
// ---------------------------------------------------------------------------
__global__ void cvt_bf16_kernel(const float* __restrict__ in,
                                unsigned short* __restrict__ out, long n4) {
    long i = (long)blockIdx.x * blockDim.x + threadIdx.x;
    if (i >= n4) return;
    float4 v = ((const float4*)in)[i];
    unsigned short r[4] = { f2bf(v.x), f2bf(v.y), f2bf(v.z), f2bf(v.w) };
    *(uint2*)(out + i * 4) = *(const uint2*)r;
}

// ---------------------------------------------------------------------------
// Generic batched bf16 GEMM:  Cz[i,j] = alpha * sum_k Az[i,k]*Bz[k,j] (+bias[j])
//   Az element (i,k) at A + z*batchA + i*sAi + k*sAk   (bf16)
//   Bz element (k,j) at B + z*batchB + k*sBk + j*sBj   (bf16)
//   Cz element (i,j) at   z*batchC + i*ldc + j         (fp32 and/or bf16 out)
// Block: 128 threads (4 waves), tile 64x64, K-step 32.
// Fast vectorized staging when sAk==1 && sBj==1 (row-major A, k-major B).
// ---------------------------------------------------------------------------
__global__ void __launch_bounds__(128)
gemm_bf16_kernel(const unsigned short* __restrict__ A,
                 const unsigned short* __restrict__ B,
                 float* __restrict__ outF,
                 unsigned short* __restrict__ outH,
                 const float* __restrict__ bias,
                 int M, int N, int K,
                 long sAi, long sAk, long sBk, long sBj,
                 long batchA, long batchB, long batchC,
                 int ldc, float alpha) {
    // 80-byte row pitch (40 ushorts): 16B aligned, conflict-free b128 gathers
    __shared__ __align__(16) unsigned short a_t[64][40];  // [i][k]
    __shared__ __align__(16) unsigned short b_t[64][40];  // [j][k]  (B^T tile)

    const int tid  = threadIdx.x;
    const int lane = tid & 31, wave = tid >> 5;
    const int wi = (wave >> 1) * 32, wj = (wave & 1) * 32;
    const int n16 = lane & 15, half = lane >> 4;
    const long i0 = (long)blockIdx.y * 64, j0 = (long)blockIdx.x * 64;
    const unsigned short* Ab = A + (long)blockIdx.z * batchA;
    const unsigned short* Bb = B + (long)blockIdx.z * batchB;
    const bool vecpath = (sAk == 1) && (sBj == 1);

    v8f acc[2][2] = {};

    for (int k0 = 0; k0 < K; k0 += 32) {
        if (vecpath) {
            // A tile: 64 rows x 32 k, contiguous rows -> b128 loads
            #pragma unroll
            for (int it = 0; it < 2; ++it) {
                int c = tid + it * 128;
                int i = c >> 2, kc = (c & 3) * 8;
                *(uint4*)&a_t[i][kc] =
                    *(const uint4*)&Ab[(i0 + i) * sAi + (k0 + kc)];
            }
            // B tile: 32 k-rows x 64 j, contiguous in j -> b128 load + scatter
            #pragma unroll
            for (int it = 0; it < 2; ++it) {
                int c = tid + it * 128;
                int k = c >> 3, jc = (c & 7) * 8;
                uint4 v = *(const uint4*)&Bb[(long)(k0 + k) * sBk + (j0 + jc)];
                const unsigned short* e = (const unsigned short*)&v;
                #pragma unroll
                for (int u = 0; u < 8; ++u) b_t[jc + u][k] = e[u];
            }
        } else {
            #pragma unroll
            for (int it = 0; it < 16; ++it) {
                int idx = tid + it * 128;
                int i = idx >> 5, k = idx & 31;
                a_t[i][k] = Ab[(i0 + i) * sAi + (long)(k0 + k) * sAk];
                b_t[i][k] = Bb[(long)(k0 + k) * sBk + (j0 + i) * sBj];
            }
        }
        __syncthreads();

        const int ka = half * 8, kb = half * 16;
        FragU af[2], bf[2];
        #pragma unroll
        for (int t = 0; t < 2; ++t) {
            af[t].v = load_frag_a(&a_t[wi + t * 16 + n16][0], ka);
            bf[t].v = load_frag_b(&b_t[wj + t * 16 + n16][0], kb);
        }
        #pragma unroll
        for (int ti = 0; ti < 2; ++ti)
            #pragma unroll
            for (int tj = 0; tj < 2; ++tj)
                acc[ti][tj] = wmma_bf16(af[ti].v, bf[tj].v, acc[ti][tj]);
        __syncthreads();
    }

    // store: C/D layout: col n = lane&15, rows m = r + 8*half
    const long cbase = (long)blockIdx.z * batchC;
    #pragma unroll
    for (int ti = 0; ti < 2; ++ti)
        #pragma unroll
        for (int tj = 0; tj < 2; ++tj) {
            long col = j0 + wj + tj * 16 + n16;
            long rbase = i0 + wi + ti * 16 + half * 8;
            float bv = bias ? bias[col] : 0.f;
            float vals[8];
            #pragma unroll
            for (int r = 0; r < 8; ++r) vals[r] = acc[ti][tj][r] * alpha + bv;
            if (outF) {
                #pragma unroll
                for (int r = 0; r < 8; ++r)
                    outF[cbase + (rbase + r) * (long)ldc + col] = vals[r];
            }
            if (outH) {
                #pragma unroll
                for (int r = 0; r < 8; ++r)
                    outH[cbase + (rbase + r) * (long)ldc + col] = f2bf(vals[r]);
            }
        }
}

// ---------------------------------------------------------------------------
// q squared norms: qsq[(t*N+b)*H + h] = sum_d Qbf[row, h*64+d]^2  (from bf16)
// ---------------------------------------------------------------------------
__global__ void qsq_kernel(const unsigned short* __restrict__ Qbf,
                           float* __restrict__ qsq) {
    int idx = blockIdx.x * blockDim.x + threadIdx.x;
    if (idx >= 8192 * 16) return;
    int row = idx >> 4, h = idx & 15;
    const unsigned short* p = Qbf + (long)row * 1024 + h * 64;
    float s = 0.f;
    #pragma unroll
    for (int i = 0; i < 64; i += 8) {
        uint4 v = *(const uint4*)(p + i);
        const unsigned short* e = (const unsigned short*)&v;
        #pragma unroll
        for (int u = 0; u < 8; ++u) { float f = bf2f(e[u]); s += f * f; }
    }
    qsq[idx] = s;
}

// ---------------------------------------------------------------------------
// Flash L2-attention, V == Q:  O[t] = softmax_s((2 q_t.q_s - |q_s|^2)/8) @ Q
// grid: (T/64, N*H); block 128 (4 waves x 16 query rows).
// K/V tiles streamed by the Tensor Data Mover (double-buffered) when present.
// ---------------------------------------------------------------------------
__global__ void __launch_bounds__(128)
flash_l2_kernel(const unsigned short* __restrict__ Qbf,
                const float* __restrict__ qsq,
                unsigned short* __restrict__ Obf) {
    __shared__ __align__(16) unsigned short q_lds[64][72];      // queries [t][d]
    __shared__ __align__(16) unsigned short kv_rm[2][32][72];   // keys    [s][d]
    __shared__ __align__(16) unsigned short kv_tr[2][64][40];   // keys^T  [d][s]
    __shared__ __align__(16) unsigned short p_lds[4][16][40];   // per-wave P tile
    __shared__ float qsq_k[2][32];

    const int tid = threadIdx.x, lane = tid & 31, wave = tid >> 5;
    const int n16 = lane & 15, half = lane >> 4;
    const int tb = blockIdx.x;
    const int b  = blockIdx.y >> 4;   // N=8
    const int h  = blockIdx.y & 15;   // H=16
    const float inv_sqrt_hd = 0.125f;
    const int ka = half * 8, kb = half * 16;

#if HAVE_TDM
    if (wave == 0) {
        // Q block: 64 rows x 64 bf16, global row stride N*D = 8192 elements
        tdm_load_2d_bf16((unsigned)(unsigned long long)(void*)&q_lds[0][0],
                         (unsigned long long)(Qbf + ((long)(tb * 64) * 8 + b) * 1024 + h * 64),
                         64, 64, 8192);
        // prime K/V buffer 0 (keys 0..31)
        tdm_load_2d_bf16((unsigned)(unsigned long long)(void*)&kv_rm[0][0][0],
                         (unsigned long long)(Qbf + (long)b * 1024 + h * 64),
                         64, 32, 8192);
    }
#else
    for (int c = tid; c < 512; c += 128) {
        int i = c >> 3, cc = c & 7;
        long row = (long)(tb * 64 + i) * 8 + b;
        *(uint4*)(&q_lds[i][cc * 8]) =
            *(const uint4*)(Qbf + row * 1024 + h * 64 + cc * 8);
    }
#endif

    v8f acc[4] = {};
    float mrun[8], lrun[8];
    #pragma unroll
    for (int r = 0; r < 8; ++r) { mrun[r] = -3.0e38f; lrun[r] = 0.f; }

    for (int it = 0; it < 32; ++it) {
        const int s0 = it * 32;
        const int buf = it & 1;

#if HAVE_TDM
        if (wave == 0) tdm_wait0();    // kv_rm[buf] (and q_lds on it==0) ready
        __syncthreads();               // ...and visible; prev compute finished
        if (wave == 0 && it + 1 < 32)  // overlap next tile's DMA with compute
            tdm_load_2d_bf16((unsigned)(unsigned long long)(void*)&kv_rm[buf ^ 1][0][0],
                             (unsigned long long)(Qbf + ((long)(s0 + 32) * 8 + b) * 1024 + h * 64),
                             64, 32, 8192);
        // build transposed copy (LDS -> LDS) for the P@V operand
        for (int c = tid; c < 256; c += 128) {
            int j = c >> 3, cc = c & 7;
            uint4 v = *(const uint4*)&kv_rm[buf][j][cc * 8];
            const unsigned short* e = (const unsigned short*)&v;
            #pragma unroll
            for (int u = 0; u < 8; ++u) kv_tr[buf][cc * 8 + u][j] = e[u];
        }
        if (tid < 32) qsq_k[buf][tid] = qsq[((long)(s0 + tid) * 8 + b) * 16 + h];
        __syncthreads();
#else
        __syncthreads();
        for (int c = tid; c < 256; c += 128) {
            int j = c >> 3, cc = c & 7;
            long row = (long)(s0 + j) * 8 + b;
            uint4 v = *(const uint4*)(Qbf + row * 1024 + h * 64 + cc * 8);
            *(uint4*)(&kv_rm[buf][j][cc * 8]) = v;
            const unsigned short* e = (const unsigned short*)&v;
            #pragma unroll
            for (int u = 0; u < 8; ++u) kv_tr[buf][cc * 8 + u][j] = e[u];
        }
        if (tid < 32) qsq_k[buf][tid] = qsq[((long)(s0 + tid) * 8 + b) * 16 + h];
        __syncthreads();
#endif

        // prefetch the tile after next into L2
        if (s0 + 64 < 1024) {
            const unsigned short* pf =
                Qbf + ((long)(s0 + 64 + (tid >> 2)) * 8 + b) * 1024 + h * 64 + (tid & 3) * 16;
            __builtin_prefetch(pf, 0, 0);
        }

        // S[m,n] = sum_d Q[m,d] * K[n,d]   (two 16-key sub-tiles, K-dim 64)
        v8f s[2] = {};
        #pragma unroll
        for (int tn = 0; tn < 2; ++tn)
            #pragma unroll
            for (int dc = 0; dc < 64; dc += 32) {
                v16bf a  = load_frag_a(&q_lds[wave * 16 + n16][dc], ka);
                v16bf bb = load_frag_b(&kv_rm[buf][tn * 16 + n16][dc], kb);
                s[tn] = wmma_bf16(a, bb, s[tn]);
            }

        // logits + online softmax in C-fragment layout
        float l0[8], l1[8], tmax[8];
        float qk0 = qsq_k[buf][n16], qk1 = qsq_k[buf][16 + n16];
        #pragma unroll
        for (int r = 0; r < 8; ++r) {
            l0[r] = (2.f * s[0][r] - qk0) * inv_sqrt_hd;
            l1[r] = (2.f * s[1][r] - qk1) * inv_sqrt_hd;
            tmax[r] = fmaxf(l0[r], l1[r]);
        }
        #pragma unroll
        for (int m = 1; m < 16; m <<= 1)
            #pragma unroll
            for (int r = 0; r < 8; ++r)
                tmax[r] = fmaxf(tmax[r], __shfl_xor(tmax[r], m, 32));

        float rsum[8], alph[8];
        #pragma unroll
        for (int r = 0; r < 8; ++r) {
            float mn = fmaxf(mrun[r], tmax[r]);
            alph[r] = __expf(mrun[r] - mn);
            mrun[r] = mn;
            l0[r] = __expf(l0[r] - mn);
            l1[r] = __expf(l1[r] - mn);
            rsum[r] = l0[r] + l1[r];
        }
        #pragma unroll
        for (int m = 1; m < 16; m <<= 1)
            #pragma unroll
            for (int r = 0; r < 8; ++r)
                rsum[r] += __shfl_xor(rsum[r], m, 32);

        #pragma unroll
        for (int r = 0; r < 8; ++r) {
            lrun[r] = lrun[r] * alph[r] + rsum[r];
            #pragma unroll
            for (int dt = 0; dt < 4; ++dt) acc[dt][r] *= alph[r];
            p_lds[wave][half * 8 + r][n16]      = f2bf(l0[r]);
            p_lds[wave][half * 8 + r][16 + n16] = f2bf(l1[r]);
        }
        // wave-local LDS write->read (cross-lane); LDS is in-order per wave
        asm volatile("s_wait_dscnt 0" ::: "memory");

        // O[m, d] += sum_s P[m,s] * V[s,d]   (V tile = kv_tr)
        #pragma unroll
        for (int dt = 0; dt < 4; ++dt) {
            v16bf a  = load_frag_a(&p_lds[wave][n16][0], ka);
            v16bf bb = load_frag_b(&kv_tr[buf][dt * 16 + n16][0], kb);
            acc[dt] = wmma_bf16(a, bb, acc[dt]);
        }
    }

    // normalize and store O (bf16), row = t*N+b, col = h*64 + d
    #pragma unroll
    for (int dt = 0; dt < 4; ++dt)
        #pragma unroll
        for (int r = 0; r < 8; ++r) {
            long t = (long)tb * 64 + wave * 16 + half * 8 + r;
            long off = (t * 8 + b) * 1024 + h * 64 + dt * 16 + n16;
            Obf[off] = f2bf(acc[dt][r] / lrun[r]);
        }
}

// ---------------------------------------------------------------------------
// Host launcher
// ---------------------------------------------------------------------------
extern "C" void kernel_launch(void* const* d_in, const int* in_sizes, int n_in,
                              void* d_out, int out_size, void* d_ws, size_t ws_size,
                              hipStream_t stream) {
    (void)in_sizes; (void)n_in; (void)out_size; (void)ws_size;
    const float* x  = (const float*)d_in[0];  // (T,N,D)   = (1024,8,1024)
    const float* qw = (const float*)d_in[1];  // (D,H,HD)  = (1024,16,64)
    const float* vw = (const float*)d_in[2];  // (D,H,HD)
    const float* ow = (const float*)d_in[3];  // (D,D)
    const float* ob = (const float*)d_in[4];  // (D,)
    float* out = (float*)d_out;               // (T,N,D) fp32

    char* w = (char*)d_ws;
    auto alloc = [&](size_t bytes) {
        void* p = (void*)w;
        w += (bytes + 255) & ~(size_t)255;
        return p;
    };
    unsigned short* x_bf  = (unsigned short*)alloc(8192l * 1024 * 2);  // reused as O
    unsigned short* wq_bf = (unsigned short*)alloc(1024l * 1024 * 2);
    unsigned short* vw_bf = (unsigned short*)alloc(1024l * 1024 * 2);
    unsigned short* ow_bf = (unsigned short*)alloc(1024l * 1024 * 2);
    unsigned short* Q_bf  = (unsigned short*)alloc(8192l * 1024 * 2);
    float*          qsq   = (float*)alloc(8192l * 16 * 4);
    unsigned short* Bh_bf = (unsigned short*)alloc(16l * 64 * 64 * 2);
    unsigned short* C_bf  = (unsigned short*)alloc(1024l * 1024 * 2);
    unsigned short* O_bf  = x_bf;  // x_bf dead after K1

    // K0: fp32 -> bf16
    cvt_bf16_kernel<<<8192 * 1024 / 4 / 256, 256, 0, stream>>>(x,  x_bf,  8192l * 1024 / 4);
    cvt_bf16_kernel<<<1024 * 1024 / 4 / 256, 256, 0, stream>>>(qw, wq_bf, 1024l * 1024 / 4);
    cvt_bf16_kernel<<<1024 * 1024 / 4 / 256, 256, 0, stream>>>(vw, vw_bf, 1024l * 1024 / 4);
    cvt_bf16_kernel<<<1024 * 1024 / 4 / 256, 256, 0, stream>>>(ow, ow_bf, 1024l * 1024 / 4);

    // K1: Q = x @ Wq   (8192x1024 x 1024x1024), bf16 out
    gemm_bf16_kernel<<<dim3(16, 128, 1), 128, 0, stream>>>(
        x_bf, wq_bf, nullptr, Q_bf, nullptr,
        8192, 1024, 1024,
        /*sAi*/1024, /*sAk*/1, /*sBk*/1024, /*sBj*/1,
        0, 0, 0, /*ldc*/1024, 1.0f);

    // K2: |q_s|^2 per (row, head)
    qsq_kernel<<<8192 * 16 / 256, 256, 0, stream>>>(Q_bf, qsq);

    // K3a: B_h = Wq_h^T @ Vw_h / sqrt(HD)   (per head, 64x64, K=1024)
    gemm_bf16_kernel<<<dim3(1, 1, 16), 128, 0, stream>>>(
        wq_bf, vw_bf, nullptr, Bh_bf, nullptr,
        64, 64, 1024,
        /*sAi*/1, /*sAk*/1024, /*sBk*/1024, /*sBj*/1,
        /*batchA*/64, /*batchB*/64, /*batchC*/4096, /*ldc*/64, 0.125f);

    // K3b: C_h = B_h @ out_w^T(head slice)   (per head, 64x1024, K=64)
    gemm_bf16_kernel<<<dim3(16, 1, 16), 128, 0, stream>>>(
        Bh_bf, ow_bf, nullptr, C_bf, nullptr,
        64, 1024, 64,
        /*sAi*/64, /*sAk*/1, /*sBk*/1, /*sBj*/1024,
        /*batchA*/4096, /*batchB*/64, /*batchC*/64l * 1024, /*ldc*/1024, 1.0f);

    // K4: flash attention (V == Q), O bf16
    flash_l2_kernel<<<dim3(16, 128), 128, 0, stream>>>(Q_bf, qsq, O_bf);

    // K5: out = O @ C + out_b   (fp32 output)
    gemm_bf16_kernel<<<dim3(16, 128, 1), 128, 0, stream>>>(
        O_bf, C_bf, out, nullptr, ob,
        8192, 1024, 1024,
        /*sAi*/1024, /*sAk*/1, /*sBk*/1024, /*sBj*/1,
        0, 0, 0, /*ldc*/1024, 1.0f);
}
